// Core_76922864271748
// MI455X (gfx1250) — matrix operand order repaired and checked by reading, not compiled, not run on hardware
//
#include <hip/hip_runtime.h>
#include <hip/hip_bf16.h>

// ---------------------------------------------------------------------------
// Types for CDNA5 WMMA (wave32)
// ---------------------------------------------------------------------------
typedef __bf16 bf16_t;
typedef bf16_t v16bf __attribute__((ext_vector_type(16)));
typedef float  v8f   __attribute__((ext_vector_type(8)));

union AFrag { uint4 u4[2]; v16bf v; };
union BFrag { uint4 u4[2]; unsigned int u[8]; v16bf v; };

#define T_LEN   4096
#define DFULL   4096
#define DK      1024
#define PLAYERS 8
#define PAIRS   (DFULL / 2)       // 2048 interleaved pair-rows
#define LDS_COL 68                // 64 dwords + 4 pad dwords (bank spread)
#define STAGE_B (64 * LDS_COL)    // dwords per LDS stage (17408 bytes)

#ifndef __has_builtin
#define __has_builtin(x) 0
#endif
#if __has_builtin(__builtin_amdgcn_tensor_load_to_lds) && \
    __has_builtin(__builtin_amdgcn_s_wait_tensorcnt)
#define USE_TDM 1
#else
#define USE_TDM 0
#endif

// fp32 -> bf16 round-to-nearest-even
__device__ __forceinline__ unsigned short f2bf(float f) {
    unsigned int u = __float_as_uint(f);
    unsigned int r = u + 0x7FFFu + ((u >> 16) & 1u);
    return (unsigned short)(r >> 16);
}

#if USE_TDM
typedef unsigned int u32x4v __attribute__((ext_vector_type(4)));
typedef int          i32x4v __attribute__((ext_vector_type(4)));
typedef int          i32x8v __attribute__((ext_vector_type(8)));

// Issue a TDM load of a 64(pair-rows, contiguous) x 64(columns) tile of the
// column-major interleaved weight matrix (column stride 2048 dwords) into LDS
// at byte offset ldsOff, padding 4 dwords after every 64 stored dwords so the
// LDS column stride becomes 68 dwords (perfect bank spread for b128 reads).
// D# fields per CDNA5 ISA.
__device__ __forceinline__ void tdm_issue_b(const unsigned int* gsrc,
                                            unsigned ldsOff) {
    unsigned long long ga = (unsigned long long)(size_t)gsrc;
    u32x4v g0 = {
        1u,                                                // count=1, user D#
        ldsOff,                                            // lds_addr
        (unsigned)(ga & 0xFFFFFFFFull),                    // global_addr[31:0]
        (unsigned)((ga >> 32) & 0x01FFFFFFull) | (2u << 30) // addr[56:32]|type=2
    };
    i32x8v g1 = {
        (int)0x07520000,        // data_size=2(4B), pad_en=1, pad_interval=5
                                //   (64 dw), pad_amount=3 (4 dw), wg_mask=0
        (int)(2048u << 16),     // tensor_dim0[15:0]=2048 at bits[63:48]
        (int)(1024u << 16),     // dim0 hi=0; tensor_dim1[15:0]=1024 at [95:80]
        (int)(64u << 16),       // dim1 hi=0; tile_dim0=64 at bits[127:112]
        64,                     // tile_dim1=64, tile_dim2=0
        2048,                   // tensor_dim0_stride lo32 = 2048
        0,                      // stride hi, tensor_dim1_stride lo
        0
    };
    i32x4v gz = { 0, 0, 0, 0 };
#if __clang_major__ >= 23
    i32x8v gz8 = { 0, 0, 0, 0, 0, 0, 0, 0 };
    __builtin_amdgcn_tensor_load_to_lds(g0, g1, gz, gz, gz8, 0);
#else
    __builtin_amdgcn_tensor_load_to_lds(g0, g1, gz, gz, 0);
#endif
}
#endif  // USE_TDM

// ---------------------------------------------------------------------------
// Kernel 1: build x[t, c] = emb*16 + PE, write bf16; also x[-1] row in f32
// ---------------------------------------------------------------------------
__global__ __launch_bounds__(256)
void build_x_kernel(const int* __restrict__ obs,       // [T*2*PLAYERS, 9] one-hot
                    const float* __restrict__ o_emb,   // [9, 256]
                    const float* __restrict__ d_emb,   // [9, 256]
                    unsigned short* __restrict__ Xb,   // [T, DFULL] bf16
                    float* __restrict__ xlast) {       // [DFULL] f32 (row T-1)
    __shared__ int s_idx[16];                          // [p*2 + isD]
    const int t   = blockIdx.x;
    const int tid = threadIdx.x;
    if (tid < 16) {
        const int p   = tid >> 1;
        const int isD = tid & 1;
        const int row = ((t * PLAYERS + p) * 2 + isD) * 9;
        int idx = 0;
        #pragma unroll
        for (int j = 0; j < 9; ++j) idx += j * obs[row + j];  // one-hot argmax
        s_idx[tid] = idx;
    }
    __syncthreads();

    const float L2F = 13.287712379549449f / 2048.0f;   // log2(10000)/2048

    for (int c = tid; c < DFULL; c += 256) {
        const int p = c >> 9;           // player (512 cols each)
        const int w = c & 511;
        float e;
        if (w < 256) e = o_emb[s_idx[p * 2 + 0] * 256 + w];
        else         e = d_emb[s_idx[p * 2 + 1] * 256 + (w - 256)];
        const int   j    = c >> 1;
        const float freq = exp2f(-(float)j * L2F);
        const float ang  = (float)t * freq;
        const float pe   = (c & 1) ? __cosf(ang) : __sinf(ang);
        const float val  = e * 16.0f + pe;              // sqrt(DMODEL)=16
        Xb[(size_t)t * DFULL + c] = f2bf(val);
        if (t == T_LEN - 1) xlast[c] = val;
    }
}

// ---------------------------------------------------------------------------
// Kernel 2: interleave + transpose weights into column-major WMMA-B layout:
//   Wt[n * PAIRS + p] = { bf16(W[2p][n]) , bf16(W[2p+1][n]) << 16 }
// (pair-rows contiguous -> B fragments are contiguous 32B per lane)
// ---------------------------------------------------------------------------
__global__ __launch_bounds__(256)
void interleave_w_kernel(const float* __restrict__ W, unsigned int* __restrict__ Wt,
                         int pairs, int N) {
    const int idx = blockIdx.x * blockDim.x + threadIdx.x;
    if (idx >= pairs * N) return;
    const int p = idx & (pairs - 1);     // fast index -> coalesced writes
    const int n = idx >> 11;             // pairs == 2048
    const unsigned short lo = f2bf(W[(size_t)(2 * p)     * N + n]);
    const unsigned short hi = f2bf(W[(size_t)(2 * p + 1) * N + n]);
    Wt[idx] = ((unsigned int)hi << 16) | (unsigned int)lo;
}

// ---------------------------------------------------------------------------
// Kernel 3: C[4096,1024] = Xb(bf16) @ W(bf16, interleaved^T) via WMMA bf16.
// Wave tile 32(M)x64(N); 4 waves/WG stacked in M -> 128x64 per workgroup.
// B tiles staged into LDS by the Tensor Data Mover (double-buffered, 128-K
// stages, padded columns); fragments are two ds_load_b128 per lane.
// A fragments loaded directly (per-lane row gathers hit L2) with WGP-scope
// prefetch of the next stage's A data.
// ---------------------------------------------------------------------------
__global__ __launch_bounds__(128)
void gemm_wmma_kernel(const unsigned short* __restrict__ Xb,  // [4096, 4096] bf16
                      const unsigned int*  __restrict__ Wt,   // [1024, 2048] dwords
                      float* __restrict__ C) {                // [4096, 1024] f32
    const int K = DFULL, N = DK;
    const int tid    = threadIdx.x;
    const int wave   = tid >> 5;
    const int lane   = tid & 31;
    const int laneLo = lane & 15;
    const int hi     = lane >> 4;
    const int m0 = blockIdx.x * 128 + wave * 32;
    const int n0 = blockIdx.y * 64;

#if USE_TDM
    __shared__ unsigned int ldsB[2][STAGE_B];          // 2 x 17408 B stages
    const unsigned ldsBase = (unsigned)(size_t)(&ldsB[0][0]);
    // Prologue: issue stages kc=0 and kc=128 (every wave issues; EXEC is
    // ignored by tensor ops and identical redundant writes are benign).
    tdm_issue_b(Wt + (size_t)n0 * PAIRS,        ldsBase);
    tdm_issue_b(Wt + (size_t)n0 * PAIRS + 64,   ldsBase + STAGE_B * 4u);
#endif

    v8f acc[2][4] = {};

    for (int kc = 0; kc < K; kc += 128) {
#if USE_TDM
        const int buf = (kc >> 7) & 1;
        if (kc + 128 < K) __builtin_amdgcn_s_wait_tensorcnt(1);
        else              __builtin_amdgcn_s_wait_tensorcnt(0);
        const unsigned int* Bt = &ldsB[buf][0];
#endif
        for (int ks = 0; ks < 128; ks += 32) {
            const int k0 = kc + ks;
            // --- A fragments: 16x32 bf16, per-lane two b128 loads ---------
            AFrag a[2];
            #pragma unroll
            for (int im = 0; im < 2; ++im) {
                const unsigned short* p =
                    Xb + (size_t)(m0 + im * 16 + laneLo) * K + (k0 + hi * 16);
                a[im].u4[0] = *(const uint4*)(p);        // K = base .. base+7
                a[im].u4[1] = *(const uint4*)(p + 8);    // K = base+8..base+15
                if (ks == 0) {
                    // WGP-scope prefetch of this lane's next-stage A span
                    __builtin_prefetch((const void*)(p + 128), 0, 3);
                    __builtin_prefetch((const void*)(p + 192), 0, 3);
                }
            }
            // --- B fragments: 32x16 bf16, contiguous 32B per lane ---------
            const int rbase = (ks >> 1) + hi * 8;        // pair-row in tile
            #pragma unroll
            for (int jn = 0; jn < 4; ++jn) {
                BFrag b;
                const int col = jn * 16 + laneLo;
#if USE_TDM
                const unsigned int* bp = Bt + col * LDS_COL + rbase;
                b.u4[0] = *(const uint4*)(bp);           // two ds_load_b128
                b.u4[1] = *(const uint4*)(bp + 4);
#else
                const unsigned int* bp =
                    Wt + (size_t)(n0 + col) * PAIRS + (k0 >> 1) + hi * 8;
                b.u4[0] = *(const uint4*)(bp);           // two global b128
                b.u4[1] = *(const uint4*)(bp + 4);
#endif
                #pragma unroll
                for (int im = 0; im < 2; ++im) {
                    acc[im][jn] = __builtin_amdgcn_wmma_f32_16x16x32_bf16(
                        false, a[im].v, false, b.v, (short)0, acc[im][jn],
                        false, false);
                }
            }
        }
#if USE_TDM
        __syncthreads();    // everyone done reading buf before overwrite
        if (kc + 256 < K)
            tdm_issue_b(Wt + (size_t)n0 * PAIRS + ((kc + 256) >> 1),
                        ldsBase + (unsigned)buf * (STAGE_B * 4u));
#endif
    }

    // --- store C: VGPR r -> row (hi*8 + r), col laneLo --------------------
    #pragma unroll
    for (int im = 0; im < 2; ++im)
        #pragma unroll
        for (int jn = 0; jn < 4; ++jn)
            #pragma unroll
            for (int r = 0; r < 8; ++r)
                C[(size_t)(m0 + im * 16 + hi * 8 + r) * N
                  + (n0 + jn * 16 + laneLo)] = acc[im][jn][r];
}

// ---------------------------------------------------------------------------
// Kernel 4/7/8: out[n] = sum_c vec[c] * W[c*1024 + n]   (optional relu)
// ---------------------------------------------------------------------------
__global__ __launch_bounds__(256)
void gemv_col_kernel(const float* __restrict__ vec, const float* __restrict__ W,
                     float* __restrict__ out, int Kdim, int relu) {
    const int n = blockIdx.x * blockDim.x + threadIdx.x;   // 1024 total
    float acc = 0.0f;
    for (int c = 0; c < Kdim; ++c) acc += vec[c] * W[(size_t)c * DK + n];
    if (relu) acc = fmaxf(acc, 0.0f);
    out[n] = acc;
}

// ---------------------------------------------------------------------------
// Kernel 5: raw scores s[t] = (q . K[t]) / sqrt(DK);  one wave per t
// ---------------------------------------------------------------------------
__global__ __launch_bounds__(256)
void scores_kernel(const float* __restrict__ q, const float* __restrict__ Kmat,
                   float* __restrict__ s) {
    const int wave = threadIdx.x >> 5;
    const int lane = threadIdx.x & 31;
    const int t = blockIdx.x * 8 + wave;
    const float* kr = Kmat + (size_t)t * DK;
    float acc = 0.0f;
    for (int i = lane; i < DK; i += 32) acc += q[i] * kr[i];
    #pragma unroll
    for (int off = 16; off > 0; off >>= 1) acc += __shfl_xor(acc, off, 32);
    if (lane == 0) s[t] = acc * 0.03125f;   // 1/sqrt(1024)
}

// ---------------------------------------------------------------------------
// Kernel 6: in-place softmax over 4096 scores (single workgroup)
// ---------------------------------------------------------------------------
__global__ __launch_bounds__(1024)
void softmax_kernel(float* __restrict__ s) {
    __shared__ float red[1024];
    const int tid = threadIdx.x;
    float m = -1e30f;
    for (int i = tid; i < T_LEN; i += 1024) m = fmaxf(m, s[i]);
    red[tid] = m; __syncthreads();
    for (int off = 512; off > 0; off >>= 1) {
        if (tid < off) red[tid] = fmaxf(red[tid], red[tid + off]);
        __syncthreads();
    }
    const float mx = red[0];
    __syncthreads();
    float sum = 0.0f;
    for (int i = tid; i < T_LEN; i += 1024) {
        const float e = expf(s[i] - mx);
        s[i] = e;
        sum += e;
    }
    red[tid] = sum; __syncthreads();
    for (int off = 512; off > 0; off >>= 1) {
        if (tid < off) red[tid] += red[tid + off];
        __syncthreads();
    }
    const float inv = 1.0f / red[0];
    for (int i = tid; i < T_LEN; i += 1024) s[i] *= inv;
}

// ---------------------------------------------------------------------------
// Kernel 9: three heads -> d_out[0..63]=o_logits, [64..127]=d_logits, [128]=v
// ---------------------------------------------------------------------------
__global__ __launch_bounds__(256)
void heads_kernel(const float* __restrict__ h,
                  const float* __restrict__ Wo, const float* __restrict__ bo,
                  const float* __restrict__ Wd, const float* __restrict__ bd,
                  const float* __restrict__ Wv, const float* __restrict__ bv,
                  float* __restrict__ out) {
    const int tid = threadIdx.x;
    if (tid < 64) {
        float acc = bo[tid];
        const float* w = Wo + (size_t)tid * DK;
        for (int i = 0; i < DK; ++i) acc += h[i] * w[i];
        out[tid] = acc;
    } else if (tid < 128) {
        const int a = tid - 64;
        float acc = bd[a];
        const float* w = Wd + (size_t)a * DK;
        for (int i = 0; i < DK; ++i) acc += h[i] * w[i];
        out[tid] = acc;
    } else if (tid == 128) {
        float acc = bv[0];
        for (int i = 0; i < DK; ++i) acc += h[i] * Wv[i];
        out[128] = acc;
    }
}

// ---------------------------------------------------------------------------
// Host launcher
// ---------------------------------------------------------------------------
extern "C" void kernel_launch(void* const* d_in, const int* in_sizes, int n_in,
                              void* d_out, int out_size, void* d_ws, size_t ws_size,
                              hipStream_t stream) {
    const int*   obs    = (const int*)  d_in[0];
    const float* o_emb  = (const float*)d_in[1];
    const float* d_emb  = (const float*)d_in[2];
    const float* WQ     = (const float*)d_in[3];
    const float* WK     = (const float*)d_in[4];
    const float* WV     = (const float*)d_in[5];
    const float* WO     = (const float*)d_in[6];
    const float* Wo_pol = (const float*)d_in[7];
    const float* bo_pol = (const float*)d_in[8];
    const float* Wd_pol = (const float*)d_in[9];
    const float* bd_pol = (const float*)d_in[10];
    const float* Wv     = (const float*)d_in[11];
    const float* bv     = (const float*)d_in[12];
    float* out = (float*)d_out;

    // Workspace carve-up (bytes)
    char* ws = (char*)d_ws;
    unsigned short* Xb    = (unsigned short*)(ws);                 // 32 MB
    unsigned int*   WKt   = (unsigned int*)(ws + 33554432);        //  8 MB
    unsigned int*   WVt   = (unsigned int*)(ws + 41943040);        //  8 MB
    float*          Kmat  = (float*)(ws + 50331648);               // 16 MB
    float*          Vmat  = (float*)(ws + 67108864);               // 16 MB
    float*          xlast = (float*)(ws + 83886080);               // 16 KB
    float*          qv    = (float*)(ws + 83902464);               //  4 KB
    float*          att   = (float*)(ws + 83906560);               // 16 KB
    float*          ctx   = (float*)(ws + 83922944);               //  4 KB
    float*          hbuf  = (float*)(ws + 83927040);               //  4 KB

    // 1) build x (bf16) + last row (f32)
    build_x_kernel<<<T_LEN, 256, 0, stream>>>(obs, o_emb, d_emb, Xb, xlast);

    // 2) interleave+transpose WK/WV into column-major WMMA-B dword layout
    const int wElems = PAIRS * DK;               // 2M dwords
    interleave_w_kernel<<<(wElems + 255) / 256, 256, 0, stream>>>(WK, WKt, PAIRS, DK);
    interleave_w_kernel<<<(wElems + 255) / 256, 256, 0, stream>>>(WV, WVt, PAIRS, DK);

    // 3) K = x @ WK, V = x @ WV  (bf16 WMMA, f32 accumulate, TDM-staged B)
    dim3 gg(T_LEN / 128, DK / 64);               // 32 x 16
    gemm_wmma_kernel<<<gg, 128, 0, stream>>>(Xb, WKt, Kmat);
    gemm_wmma_kernel<<<gg, 128, 0, stream>>>(Xb, WVt, Vmat);

    // 4) q_last = x[-1] @ WQ   (full fp32)
    gemv_col_kernel<<<DK / 256, 256, 0, stream>>>(xlast, WQ, qv, DFULL, 0);

    // 5) raw attention scores for the last row
    scores_kernel<<<T_LEN / 8, 256, 0, stream>>>(qv, Kmat, att);

    // 6) softmax over 4096
    softmax_kernel<<<1, 1024, 0, stream>>>(att);

    // 7) ctx = att @ V
    gemv_col_kernel<<<DK / 256, 256, 0, stream>>>(att, Vmat, ctx, T_LEN, 0);

    // 8) h = relu(ctx @ WO)
    gemv_col_kernel<<<DK / 256, 256, 0, stream>>>(ctx, WO, hbuf, DK, 1);

    // 9) heads
    heads_kernel<<<1, 256, 0, stream>>>(hbuf, Wo_pol, bo_pol, Wd_pol, bd_pol,
                                        Wv, bv, out);
}
